// Attentionv2_29901562315050
// MI455X (gfx1250) — matrix operand order repaired and checked by reading, not compiled, not run on hardware
//
#include <hip/hip_runtime.h>

// ---------------------------------------------------------------------------
// Problem dimensions (fixed by the reference's setup_inputs)
// ---------------------------------------------------------------------------
constexpr int kT = 128, kS = 128, kB = 32, kH = 512, kC = 512, kM = 256;

typedef __attribute__((ext_vector_type(16))) __bf16 v16bf;
typedef __attribute__((ext_vector_type(8)))  float  v8f;

union BfTile {
  v16bf v;
  uint4 q[2];
};

__device__ __forceinline__ unsigned short bf_bits(float f) {
  unsigned u = __builtin_bit_cast(unsigned, f);
  u += 0x7fffu + ((u >> 16) & 1u);          // round-to-nearest-even
  return (unsigned short)(u >> 16);
}

__device__ __forceinline__ float fast_tanh(float x) {
  // tanh(x) = (e^{2x}-1)/(e^{2x}+1); clamp so e never overflows.
  float xc = fminf(fmaxf(x, -15.0f), 15.0f);
  float e  = __expf(xc * 2.0f);
  return (e - 1.0f) * __builtin_amdgcn_rcpf(e + 1.0f);
}

// ---------------------------------------------------------------------------
// f32 -> bf16 bulk convert (4 elements / thread, b128 in, b64 out)
// ---------------------------------------------------------------------------
__global__ __launch_bounds__(256) void cvt_f32_to_bf16_vec4(
    const float4* __restrict__ in, ushort4* __restrict__ out, int n4) {
  int i = blockIdx.x * blockDim.x + threadIdx.x;
  if (i >= n4) return;
  float4 v = in[i];
  ushort4 h;
  h.x = bf_bits(v.x); h.y = bf_bits(v.y); h.z = bf_bits(v.z); h.w = bf_bits(v.w);
  out[i] = h;
}

// ctx (S,B,C) f32  ->  ctxT (B,C,S) bf16  (makes attended-GEMM B operand
// K(=s)-contiguous so WMMA B loads are single 32B chunks)
__global__ __launch_bounds__(256) void cvt_ctx_transpose(
    const float* __restrict__ ctx, unsigned short* __restrict__ ctxT) {
  int idx = blockIdx.x * blockDim.x + threadIdx.x;
  if (idx >= kS * kB * kC) return;
  int c = idx % kC;
  int b = (idx / kC) % kB;
  int s = idx / (kC * kB);
  ctxT[((size_t)b * kC + c) * kS + s] = bf_bits(ctx[idx]);
}

// ---------------------------------------------------------------------------
// Generic GEMM:  Y(rows x N, f32) = X(rows x K, bf16) * W(N x K, bf16)^T
// One wave per 16x64 output strip (4 N-tiles, 4 f32 accumulators): the A
// fragment is loaded once per 32-deep k-step and reused by 4 WMMAs
// (2.5 b128 loads / WMMA instead of 4).
//   A layout: lane l -> row (l&15), K chunk  at (l>>4)*16     (2 x 16B)
//   B layout: lane l -> col (l&15), K chunk  at (l>>4)*16     (1 x 32B)
//   D layout: VGPR v -> M=(l>>4)*8+v, N=(l&15)
// ---------------------------------------------------------------------------
__global__ __launch_bounds__(256) void gemm_bf16_xwT(
    const unsigned short* __restrict__ X, const unsigned short* __restrict__ W,
    float* __restrict__ Y, int rows, int K, int N) {
  const int lane   = threadIdx.x & 31;
  const int wave   = threadIdx.x >> 5;
  const int strN   = N >> 6;                          // 16x64 strips along N
  const int strip  = blockIdx.x * (blockDim.x >> 5) + wave;
  if (strip >= (rows >> 4) * strN) return;            // wave-uniform guard
  const int tm  = strip / strN;
  const int tn0 = (strip % strN) << 2;                // first of 4 N-tiles
  const int r   = (tm << 4) + (lane & 15);
  const int n   = (tn0 << 4) + (lane & 15);
  const int kbA = (lane >> 4) << 4;                   // 0 or 16
  const int kbB = (lane >> 4) << 4;                   // 0 or 16
  const unsigned short* pa = X + (size_t)r * K + kbA;
  const unsigned short* pb = W + (size_t)n * K + kbB;
  const size_t wstride = (size_t)16 * K;              // 16 W-rows per N-tile
  v8f acc0 = {}, acc1 = {}, acc2 = {}, acc3 = {};
  for (int k0 = 0; k0 < K; k0 += 32) {
    BfTile A, B0, B1, B2, B3;
    A.q[0]  = *(const uint4*)(pa + k0);
    A.q[1]  = *(const uint4*)(pa + k0 + 8);
    B0.q[0] = *(const uint4*)(pb + k0);
    B0.q[1] = *(const uint4*)(pb + k0 + 8);
    B1.q[0] = *(const uint4*)(pb + wstride + k0);
    B1.q[1] = *(const uint4*)(pb + wstride + k0 + 8);
    B2.q[0] = *(const uint4*)(pb + 2 * wstride + k0);
    B2.q[1] = *(const uint4*)(pb + 2 * wstride + k0 + 8);
    B3.q[0] = *(const uint4*)(pb + 3 * wstride + k0);
    B3.q[1] = *(const uint4*)(pb + 3 * wstride + k0 + 8);
    acc0 = __builtin_amdgcn_wmma_f32_16x16x32_bf16(
        false, A.v, false, B0.v, (short)0, acc0, false, false);
    acc1 = __builtin_amdgcn_wmma_f32_16x16x32_bf16(
        false, A.v, false, B1.v, (short)0, acc1, false, false);
    acc2 = __builtin_amdgcn_wmma_f32_16x16x32_bf16(
        false, A.v, false, B2.v, (short)0, acc2, false, false);
    acc3 = __builtin_amdgcn_wmma_f32_16x16x32_bf16(
        false, A.v, false, B3.v, (short)0, acc3, false, false);
  }
  const int m0 = (lane >> 4) << 3;
  float* py = Y + (size_t)((tm << 4) + m0) * N + (tn0 << 4) + (lane & 15);
#pragma unroll
  for (int v = 0; v < 8; ++v) {
    float* p = py + (size_t)v * N;
    p[0]  = acc0[v];
    p[16] = acc1[v];
    p[32] = acc2[v];
    p[48] = acc3[v];
  }
}

// ---------------------------------------------------------------------------
// Fused scores + softmax.  Block = (one batch b, 32 t-columns), 256 threads.
// score[s,t] = sum_m w_ff[m] * tanh(hid_m[t,b,m] + ctx_m[s,b,m])
// LDS-blocked over m in chunks of 64; each thread owns 16 (s,t) pairs and
// accumulates in registers.  Then masked softmax over s, writing both the
// final (T,S,B) alpha output and a bf16 (B,T,S) copy for the next GEMM.
// LDS: 32KB ctx-chunk + 8.1KB hid-chunk(pad 65) + 16.5KB scores(pad 33) ~ 58KB
// ---------------------------------------------------------------------------
__global__ __launch_bounds__(256) void scores_softmax_kernel(
    const float* __restrict__ hid_m,      // (T*B, M)
    const float* __restrict__ ctx_m,      // (S*B, M)
    const float* __restrict__ w_ff,       // (M)
    const float* __restrict__ mask,       // (S,B)
    float* __restrict__ out_alpha,        // (T,S,B)  -> d_out region 0
    unsigned short* __restrict__ alpha_bts) {  // (B,T,S) bf16
  __shared__ float s_ctx[kS * 64];
  __shared__ float s_hid[32 * 65];
  __shared__ float s_wff[64];
  __shared__ float s_sc[kS * 33];
  __shared__ float s_mx[32];
  __shared__ float s_ri[32];
  const int b   = blockIdx.y;
  const int t0  = blockIdx.x * 32;
  const int tid = threadIdx.x;
  const int tl  = tid & 31;   // local t
  const int sr  = tid >> 5;   // s residue 0..7
  float acc[16];
#pragma unroll
  for (int i = 0; i < 16; ++i) acc[i] = 0.0f;

  for (int m0 = 0; m0 < kM; m0 += 64) {
    __syncthreads();
    for (int i = tid; i < kS * 64; i += 256) {
      int s = i >> 6, m = i & 63;
      s_ctx[i] = ctx_m[((size_t)s * kB + b) * kM + m0 + m];
    }
    for (int i = tid; i < 32 * 64; i += 256) {
      int t = i >> 6, m = i & 63;
      s_hid[t * 65 + m] = hid_m[((size_t)(t0 + t) * kB + b) * kM + m0 + m];
    }
    if (tid < 64) s_wff[tid] = w_ff[m0 + tid];
    __syncthreads();
#pragma unroll 1
    for (int i = 0; i < 16; ++i) {
      const int s = sr + (i << 3);
      const float* pc = &s_ctx[s << 6];     // broadcast across wave (same s)
      const float* ph = &s_hid[tl * 65];    // pad-65 -> conflict-free
      float a = acc[i];
#pragma unroll 8
      for (int m = 0; m < 64; ++m)
        a = fmaf(s_wff[m], fast_tanh(ph[m] + pc[m]), a);
      acc[i] = a;
    }
  }
  __syncthreads();
#pragma unroll
  for (int i = 0; i < 16; ++i) {
    const int s = sr + (i << 3);
    float sc = acc[i];                       // TEMP == 1.0
    if (!(mask[(size_t)s * kB + b] > 0.0f)) sc = -1.0e8f;
    s_sc[s * 33 + tl] = sc;
  }
  __syncthreads();
  if (tid < 32) {                            // wave 0, no intra-wave divergence
    float mx = -3.4e38f;
    for (int s = 0; s < kS; ++s) mx = fmaxf(mx, s_sc[s * 33 + tid]);
    float sum = 0.0f;
    for (int s = 0; s < kS; ++s) sum += __expf(s_sc[s * 33 + tid] - mx);
    s_mx[tid] = mx;
    s_ri[tid] = __builtin_amdgcn_rcpf(sum);
  }
  __syncthreads();
#pragma unroll
  for (int i = 0; i < 16; ++i) {
    const int s = sr + (i << 3);
    const int t = t0 + tl;
    float al = __expf(s_sc[s * 33 + tl] - s_mx[tl]) * s_ri[tl];
    out_alpha[((size_t)t * kS + s) * kB + b] = al;
    alpha_bts[((size_t)b * kT + t) * kS + s] = bf_bits(al);
  }
}

// ---------------------------------------------------------------------------
// attended(t,b,c) = sum_s alpha[s,b,t] * ctx[s,b,c]
// Batched WMMA GEMM: per batch b, A = alpha_bts[b] (T x S), B = ctxT[b]
// ((B,C,S) layout -> K-contiguous).  One wave per 16x64 strip (4 C-tiles),
// sharing the A fragment across 4 WMMAs per k-step.  Output bf16 (T,B,C)
// since it only feeds the final bf16 GEMM.
// ---------------------------------------------------------------------------
__global__ __launch_bounds__(256) void attended_gemm(
    const unsigned short* __restrict__ alpha_bts,  // (B,T,S) bf16
    const unsigned short* __restrict__ ctxT,       // (B,C,S) bf16
    unsigned short* __restrict__ att) {            // (T,B,C) bf16
  const int lane  = threadIdx.x & 31;
  const int wave  = threadIdx.x >> 5;
  const int strC  = kC >> 6;                       // 8 strips of 64 cols
  const int tilesT = kT >> 4;                      // 8
  const int strip = blockIdx.x * (blockDim.x >> 5) + wave;
  if (strip >= kB * tilesT * strC) return;         // wave-uniform guard
  const int b   = strip / (strC * tilesT);
  const int rem = strip % (strC * tilesT);
  const int tt  = rem / strC;
  const int tc0 = (rem % strC) << 2;               // first of 4 C-tiles
  const int r   = (tt << 4) + (lane & 15);
  const int n   = (tc0 << 4) + (lane & 15);
  const int kbA = (lane >> 4) << 4;
  const int kbB = (lane >> 4) << 4;
  const unsigned short* pa = alpha_bts + ((size_t)b * kT + r) * kS + kbA;
  const unsigned short* pb = ctxT      + ((size_t)b * kC + n) * kS + kbB;
  const size_t cstride = (size_t)16 * kS;          // 16 ctxT-rows per C-tile
  v8f acc0 = {}, acc1 = {}, acc2 = {}, acc3 = {};
#pragma unroll
  for (int s0 = 0; s0 < kS; s0 += 32) {
    BfTile A, B0, B1, B2, B3;
    A.q[0]  = *(const uint4*)(pa + s0);
    A.q[1]  = *(const uint4*)(pa + s0 + 8);
    B0.q[0] = *(const uint4*)(pb + s0);
    B0.q[1] = *(const uint4*)(pb + s0 + 8);
    B1.q[0] = *(const uint4*)(pb + cstride + s0);
    B1.q[1] = *(const uint4*)(pb + cstride + s0 + 8);
    B2.q[0] = *(const uint4*)(pb + 2 * cstride + s0);
    B2.q[1] = *(const uint4*)(pb + 2 * cstride + s0 + 8);
    B3.q[0] = *(const uint4*)(pb + 3 * cstride + s0);
    B3.q[1] = *(const uint4*)(pb + 3 * cstride + s0 + 8);
    acc0 = __builtin_amdgcn_wmma_f32_16x16x32_bf16(
        false, A.v, false, B0.v, (short)0, acc0, false, false);
    acc1 = __builtin_amdgcn_wmma_f32_16x16x32_bf16(
        false, A.v, false, B1.v, (short)0, acc1, false, false);
    acc2 = __builtin_amdgcn_wmma_f32_16x16x32_bf16(
        false, A.v, false, B2.v, (short)0, acc2, false, false);
    acc3 = __builtin_amdgcn_wmma_f32_16x16x32_bf16(
        false, A.v, false, B3.v, (short)0, acc3, false, false);
  }
  const int m0 = (lane >> 4) << 3;
  const int cc = (tc0 << 4) + (lane & 15);
#pragma unroll
  for (int v = 0; v < 8; ++v) {
    unsigned short* p =
        att + ((size_t)((tt << 4) + m0 + v) * kB + b) * kC + cc;
    p[0]  = bf_bits(acc0[v]);
    p[16] = bf_bits(acc1[v]);
    p[32] = bf_bits(acc2[v]);
    p[48] = bf_bits(acc3[v]);
  }
}

// ---------------------------------------------------------------------------
// Host-side orchestration
// ---------------------------------------------------------------------------
extern "C" void kernel_launch(void* const* d_in, const int* in_sizes, int n_in,
                              void* d_out, int out_size, void* d_ws, size_t ws_size,
                              hipStream_t stream) {
  (void)in_sizes; (void)n_in; (void)out_size; (void)ws_size;
  const float* hid  = (const float*)d_in[0];   // (T,B,H)
  const float* ctx  = (const float*)d_in[1];   // (S,B,C)
  const float* mask = (const float*)d_in[2];   // (S,B)
  const float* Whid = (const float*)d_in[3];   // (M,H)
  const float* Wctx = (const float*)d_in[4];   // (M,C)
  const float* wff  = (const float*)d_in[5];   // (M)
  const float* Watt = (const float*)d_in[6];   // (H,C)

  char* ws = (char*)d_ws;
  size_t off = 0;
  auto alloc = [&](size_t bytes) -> char* {
    char* p = ws + off;
    off += (bytes + 255) & ~size_t(255);
    return p;
  };
  unsigned short* hid_bf   = (unsigned short*)alloc((size_t)kT * kB * kH * 2);
  unsigned short* ctx_bf   = (unsigned short*)alloc((size_t)kS * kB * kC * 2);
  unsigned short* Whid_bf  = (unsigned short*)alloc((size_t)kM * kH * 2);
  unsigned short* Wctx_bf  = (unsigned short*)alloc((size_t)kM * kC * 2);
  unsigned short* Watt_bf  = (unsigned short*)alloc((size_t)kH * kC * 2);
  unsigned short* ctxT_bf  = (unsigned short*)alloc((size_t)kB * kC * kS * 2);
  float*          hid_m    = (float*)alloc((size_t)kT * kB * kM * 4);
  float*          ctx_m    = (float*)alloc((size_t)kS * kB * kM * 4);
  unsigned short* alpha_bts= (unsigned short*)alloc((size_t)kB * kT * kS * 2);
  unsigned short* att_bf   = (unsigned short*)alloc((size_t)kT * kB * kC * 2);

  float* out_alpha = (float*)d_out;                           // (T,S,B)
  float* out_z     = (float*)d_out + (size_t)kT * kS * kB;    // (T,B,H)

  // --- bf16 conversion passes (bandwidth-trivial; keeps WMMA loops pure) ---
  auto cvt = [&](const float* in, unsigned short* out, int n) {
    int n4 = n / 4;
    cvt_f32_to_bf16_vec4<<<(n4 + 255) / 256, 256, 0, stream>>>(
        (const float4*)in, (ushort4*)out, n4);
  };
  cvt(hid,  hid_bf,  kT * kB * kH);
  cvt(ctx,  ctx_bf,  kS * kB * kC);
  cvt(Whid, Whid_bf, kM * kH);
  cvt(Wctx, Wctx_bf, kM * kC);
  cvt(Watt, Watt_bf, kH * kC);
  cvt_ctx_transpose<<<(kS * kB * kC + 255) / 256, 256, 0, stream>>>(ctx, ctxT_bf);

  // --- stage 1: mid projections (WMMA, 16x64 strips) ---
  {
    int strips = (kT * kB / 16) * (kM / 64);   // 1024
    gemm_bf16_xwT<<<strips / 8, 256, 0, stream>>>(hid_bf, Whid_bf, hid_m,
                                                  kT * kB, kH, kM);
    gemm_bf16_xwT<<<strips / 8, 256, 0, stream>>>(ctx_bf, Wctx_bf, ctx_m,
                                                  kS * kB, kC, kM);
  }

  // --- stage 2: tanh scores + masked softmax (VALU/trans-bound) ---
  {
    dim3 grid(kT / 32, kB);
    scores_softmax_kernel<<<grid, 256, 0, stream>>>(hid_m, ctx_m, wff, mask,
                                                    out_alpha, alpha_bts);
  }

  // --- stage 3: attended = alpha^T @ ctx, batched over b (WMMA) ---
  {
    int strips = kB * (kT / 16) * (kC / 64);   // 2048
    attended_gemm<<<strips / 8, 256, 0, stream>>>(alpha_bts, ctxT_bf, att_bf);
  }

  // --- stage 4: z = attended @ W_att2hid^T (WMMA, 16x64 strips) ---
  {
    int strips = (kT * kB / 16) * (kH / 64);   // 2048
    gemm_bf16_xwT<<<strips / 8, 256, 0, stream>>>(att_bf, Watt_bf, out_z,
                                                  kT * kB, kC, kH);
  }
}
